// DLRM_15290083574229
// MI455X (gfx1250) — matrix operand order repaired and checked by reading, not compiled, not run on hardware
//
#include <hip/hip_runtime.h>
#include <hip/hip_bf16.h>
#include <math.h>

#define B_TOTAL   32768
#define NUM_SPARSE 26
#define EMB       128
#define DENSE_IN  13
#define VOCAB     100000
#define N_FIELDS  27      // 1 + NUM_SPARSE
#define TOP_IN    479     // EMB + 351
#define KP        480     // padded top-input width (multiple of 32)

typedef __bf16 bf16;
typedef __attribute__((ext_vector_type(16))) __bf16 v16bf;
typedef __attribute__((ext_vector_type(8)))  float  v8f;

union BF16x16 { v16bf v; bf16 h[16]; uint4 q[2]; };
union BF16x4  { bf16 h[4]; uint2 u; };
union F32x8   { v8f v; float f[8]; };

// ---- WMMA 16x16x32 bf16 fragment maps (ISA 7.12.2) ----
// A (16x32): lane holds row M=lane&15; this code packs element j -> K = j + 16*(lane>=16) (contiguous, not ISA table order)
// B (32x16): lane holds col N=lane&15; this code packs element j -> K = j + 16*(lane>=16) (contiguous, not ISA table order)
// C (16x16 f32): lane holds col N=lane&15; reg r -> M = r + 8*(lane>=16)

// inverse A map: (row m, k) -> lane' and j (element slot in packed fragment)
__device__ __forceinline__ int a_lane(int m, int k) { return (m & 15) + 16 * ((k >> 4) & 1); }
__device__ __forceinline__ int a_j(int k)           { return k & 15; }
// inverse B map: (col n, k) -> lane' and j
__device__ __forceinline__ int b_lane(int n, int k) { return (n & 15) + 16 * ((k >> 4) & 1); }
__device__ __forceinline__ int b_j(int k)           { return k & 15; }

__device__ __forceinline__ v8f wmma_bf16(const BF16x16& a, const BF16x16& b, v8f c) {
  return __builtin_amdgcn_wmma_f32_16x16x32_bf16(false, a.v, false, b.v,
                                                 (short)0, c, false, false);
}
__device__ __forceinline__ BF16x16 load_frag(const bf16* p) {   // 32B = 2 x b128
  BF16x16 r;
  r.q[0] = ((const uint4*)p)[0];
  r.q[1] = ((const uint4*)p)[1];
  return r;
}

// -------- Kernel 0: pack weights to bf16 B-fragment order (one-time, tiny) --------
// Fragment array layout: [ntile][ktile][lane(32)][j(16)] ; element = W[k][n],
// n = nt*16 + (lane&15), k = kt*32 + j + 16*(lane>>4)
__global__ void dlrm_pack_weights(const float* __restrict__ bw1, const float* __restrict__ bw2,
                                  const float* __restrict__ tw1, const float* __restrict__ tw2,
                                  bf16* __restrict__ W1f, bf16* __restrict__ W2f,
                                  bf16* __restrict__ T1f, bf16* __restrict__ T2f) {
  const int stride = gridDim.x * blockDim.x;
  const int t = blockIdx.x * blockDim.x + threadIdx.x;
  for (int i = t; i < 16 * 512; i += stride) {              // bw1: 13(->32) x 256
    int nt = i >> 9, lane = (i >> 4) & 31, j = i & 15;
    int n = nt * 16 + (lane & 15), k = j + 16 * (lane >> 4);
    W1f[i] = (bf16)((k < DENSE_IN) ? bw1[k * 256 + n] : 0.f);
  }
  for (int i = t; i < 8 * 8 * 512; i += stride) {           // bw2: 256 x 128
    int nt = i >> 12, kt = (i >> 9) & 7, lane = (i >> 4) & 31, j = i & 15;
    int n = nt * 16 + (lane & 15), k = kt * 32 + j + 16 * (lane >> 4);
    W2f[i] = (bf16)bw2[k * 128 + n];
  }
  for (int i = t; i < 8 * 15 * 512; i += stride) {          // tw1: 479(->480) x 128
    int nt = i / (15 * 512), rem = i % (15 * 512);
    int kt = rem >> 9, lane = (rem >> 4) & 31, j = rem & 15;
    int n = nt * 16 + (lane & 15), k = kt * 32 + j + 16 * (lane >> 4);
    T1f[i] = (bf16)((k < TOP_IN) ? tw1[k * 128 + n] : 0.f);
  }
  for (int i = t; i < 4 * 4 * 512; i += stride) {           // tw2: 128 x 64
    int nt = i >> 11, kt = (i >> 9) & 3, lane = (i >> 4) & 31, j = i & 15;
    int n = nt * 16 + (lane & 15), k = kt * 32 + j + 16 * (lane >> 4);
    T2f[i] = (bf16)tw2[k * 64 + n];
  }
}

// ---------------- Kernel 1: bottom MLP 13 -> 256 -> 128 (relu) ----------------
__global__ __launch_bounds__(128) void dlrm_bottom_mlp(
    const float* __restrict__ dense_x, const bf16* __restrict__ W1f,
    const float* __restrict__ bb1, const bf16* __restrict__ W2f,
    const float* __restrict__ bb2, bf16* __restrict__ xfinal) {
  __shared__ __align__(16) bf16 h_lds[4][8 * 512];   // A-packed h, 8KB/wave
  const int wave = threadIdx.x >> 5, lane = threadIdx.x & 31;
  const int rowBase = (blockIdx.x * 4 + wave) * 16;
  const int mlo = lane & 15;
  const int mhi = (lane & 16) ? 8 : 0;

  // A fragment for layer 1 (K padded 13 -> 32): direct per-element build
  BF16x16 a1;
#pragma unroll
  for (int j = 0; j < 16; ++j) {
    int k = j + ((lane & 16) ? 16 : 0);
    a1.h[j] = (bf16)((k < DENSE_IN) ? dense_x[(size_t)(rowBase + mlo) * DENSE_IN + k] : 0.f);
  }
  // layer 1: 16 N-tiles; relu; stage to LDS in A-fragment order for layer 2
  for (int nt = 0; nt < 16; ++nt) {
    const int n = nt * 16 + mlo;
    BF16x16 b = load_frag(W1f + nt * 512 + lane * 16);
    F32x8 c;
#pragma unroll
    for (int r = 0; r < 8; ++r) c.f[r] = 0.f;
    c.v = wmma_bf16(a1, b, c.v);
    const float bias = bb1[n];
#pragma unroll
    for (int r = 0; r < 8; ++r) {
      float h = c.f[r] + bias;
      h = h > 0.f ? h : 0.f;
      // h[m][n] becomes A element (row m, K=n) of layer 2
      h_lds[wave][(n >> 5) * 512 + a_lane(r + mhi, n) * 16 + a_j(n)] = (bf16)h;
    }
  }
  // layer 2: K=256 (8 steps), N=128 (8 tiles) -- all-b128 fragment loads
  for (int nt = 0; nt < 8; ++nt) {
    const int n = nt * 16 + mlo;
    F32x8 c;
    const float bias = bb2[n];
#pragma unroll
    for (int r = 0; r < 8; ++r) c.f[r] = bias;
#pragma unroll
    for (int kt = 0; kt < 8; ++kt) {
      BF16x16 a = load_frag(&h_lds[wave][kt * 512 + lane * 16]);
      BF16x16 b = load_frag(W2f + (nt * 8 + kt) * 512 + lane * 16);
      c.v = wmma_bf16(a, b, c.v);
    }
#pragma unroll
    for (int r = 0; r < 8; ++r) {
      float v = c.f[r];
      xfinal[(size_t)(rowBase + r + mhi) * KP + n] = (bf16)(v > 0.f ? v : 0.f);
    }
  }
}

// ------- Kernel 2: embedding gather + pairwise interactions (WMMA) -------
// One sample per wave; 2 waves/block. Gathered rows scattered into BOTH
// A-packed and B-packed LDS fragment stores (packed b64 stores).
__global__ __launch_bounds__(64) void dlrm_interact(
    const int* __restrict__ sparse_x, const float* __restrict__ emb_tables,
    bf16* __restrict__ xfinal) {
  __shared__ __align__(16) bf16 combA[2][2 * 4 * 512];   // [tile][kt][lane][16]
  __shared__ __align__(16) bf16 combB[2][2 * 4 * 512];
  const int wave = threadIdx.x >> 5, lane = threadIdx.x & 31;
  const int sample = blockIdx.x * 2 + wave;
  const int mlo = lane & 15;
  const int mhi = (lane & 16) ? 8 : 0;
  const int k0 = lane * 4;                               // this lane's 4 columns
  const int kt0 = k0 >> 5;
  const int jA = a_j(k0), jB = b_j(k0);

  auto scatter = [&](int f, BF16x4 pk) {                 // row f, cols k0..k0+3
    const int tile = (f >> 4) * 2048;
    *(uint2*)&combA[wave][tile + kt0 * 512 + a_lane(f, k0) * 16 + jA] = pk.u;
    *(uint2*)&combB[wave][tile + kt0 * 512 + b_lane(f, k0) * 16 + jB] = pk.u;
  };

  // field 0: x_bottom, already bf16 in x_final
  {
    BF16x4 pk;
    pk.u = *(const uint2*)&xfinal[(size_t)sample * KP + k0];
    scatter(0, pk);
  }
  // fields 1..26: gather 512B rows (global_load_b128/lane), cvt, scatter
  for (int f = 0; f < NUM_SPARSE; ++f) {
    const int idx = sparse_x[(size_t)sample * NUM_SPARSE + f];
    const float4* row = (const float4*)(emb_tables + ((size_t)f * VOCAB + (size_t)idx) * EMB);
    if (f + 1 < NUM_SPARSE) {
      const int nidx = sparse_x[(size_t)sample * NUM_SPARSE + f + 1];
      __builtin_prefetch(emb_tables + ((size_t)(f + 1) * VOCAB + (size_t)nidx) * EMB, 0, 1);
    }
    float4 v = row[lane];
    BF16x4 pk;
    pk.h[0] = (bf16)v.x; pk.h[1] = (bf16)v.y; pk.h[2] = (bf16)v.z; pk.h[3] = (bf16)v.w;
    scatter(f + 1, pk);
  }
  // zero-pad fields 27..31
  for (int f = N_FIELDS; f < 32; ++f) {
    BF16x4 z; z.u = make_uint2(0u, 0u);
    scatter(f, z);
  }

  // interactions: 2x2 tiles of 16x16, K = 4 x 32 -> 16 WMMAs, pure b128 feeds
  F32x8 c00, c01, c10, c11;
#pragma unroll
  for (int r = 0; r < 8; ++r) { c00.f[r] = c01.f[r] = c10.f[r] = c11.f[r] = 0.f; }
#pragma unroll
  for (int kt = 0; kt < 4; ++kt) {
    BF16x16 a0 = load_frag(&combA[wave][kt * 512 + lane * 16]);
    BF16x16 a1 = load_frag(&combA[wave][2048 + kt * 512 + lane * 16]);
    BF16x16 b0 = load_frag(&combB[wave][kt * 512 + lane * 16]);
    BF16x16 b1 = load_frag(&combB[wave][2048 + kt * 512 + lane * 16]);
    c00.v = wmma_bf16(a0, b0, c00.v);
    c01.v = wmma_bf16(a0, b1, c01.v);
    c10.v = wmma_bf16(a1, b0, c10.v);
    c11.v = wmma_bf16(a1, b1, c11.v);
  }

  bf16* xi = xfinal + (size_t)sample * KP + EMB;
  auto emit = [&](float val, int f, int g) {
    if (f < N_FIELDS && g < N_FIELDS && f < g) {
      int idx = 26 * f - (f * (f - 1)) / 2 + (g - f - 1);  // triu(k=1) flat index
      xi[idx] = (bf16)val;
    }
  };
#pragma unroll
  for (int r = 0; r < 8; ++r) {
    int fr = r + mhi;
    emit(c00.f[r], fr,      mlo);
    emit(c01.f[r], fr,      mlo + 16);
    emit(c10.f[r], fr + 16, mlo);
    emit(c11.f[r], fr + 16, mlo + 16);
  }
  if (lane == 0) xi[TOP_IN - EMB] = (bf16)0.f;  // pad column 479
}

// ---------------- Kernel 3: top MLP 479 -> 128 -> 64 -> 1, sigmoid ----------------
__global__ __launch_bounds__(64) void dlrm_top_mlp(
    const bf16* __restrict__ xfinal,
    const bf16* __restrict__ T1f, const float* __restrict__ tb1,
    const bf16* __restrict__ T2f, const float* __restrict__ tb2,
    const float* __restrict__ tw3, const float* __restrict__ tb3,
    float* __restrict__ out) {
  __shared__ __align__(16) bf16 xlA[2][15 * 512];   // A-packed x_final rows, 15KB/wave
  __shared__ __align__(16) bf16 t1A[2][4 * 512];    // A-packed t1, 4KB/wave
  __shared__ __align__(16) bf16 t2r[2][16 * 64];    // row-major t2, 2KB/wave
  const int wave = threadIdx.x >> 5, lane = threadIdx.x & 31;
  const int rowBase = (blockIdx.x * 2 + wave) * 16;
  const int mlo = lane & 15;
  const int mhi = (lane & 16) ? 8 : 0;

  // stage 16 contiguous x_final rows into A-fragment order (coalesced b64 reads,
  // packed b64 LDS stores: 480 % 4 == 0 so each uint2 stays in one 8-col group)
  {
    const bf16* src = xfinal + (size_t)rowBase * KP;
    for (int i = lane * 4; i < 16 * KP; i += 32 * 4) {
      int m = i / KP, k = i % KP;
      uint2 w = *(const uint2*)&src[i];
      *(uint2*)&xlA[wave][(k >> 5) * 512 + a_lane(m, k) * 16 + a_j(k)] = w;
    }
  }
  // layer 1: K=480(479 valid), N=128
  for (int nt = 0; nt < 8; ++nt) {
    const int n = nt * 16 + mlo;
    F32x8 c;
    const float bias = tb1[n];
#pragma unroll
    for (int r = 0; r < 8; ++r) c.f[r] = bias;
#pragma unroll
    for (int kt = 0; kt < 15; ++kt) {
      BF16x16 a = load_frag(&xlA[wave][kt * 512 + lane * 16]);
      BF16x16 b = load_frag(T1f + ((size_t)nt * 15 + kt) * 512 + lane * 16);
      c.v = wmma_bf16(a, b, c.v);
    }
#pragma unroll
    for (int r = 0; r < 8; ++r) {
      float v = c.f[r];
      v = v > 0.f ? v : 0.f;
      t1A[wave][(n >> 5) * 512 + a_lane(r + mhi, n) * 16 + a_j(n)] = (bf16)v;
    }
  }
  // layer 2: K=128, N=64
  for (int nt = 0; nt < 4; ++nt) {
    const int n = nt * 16 + mlo;
    F32x8 c;
    const float bias = tb2[n];
#pragma unroll
    for (int r = 0; r < 8; ++r) c.f[r] = bias;
#pragma unroll
    for (int kt = 0; kt < 4; ++kt) {
      BF16x16 a = load_frag(&t1A[wave][kt * 512 + lane * 16]);
      BF16x16 b = load_frag(T2f + (nt * 4 + kt) * 512 + lane * 16);
      c.v = wmma_bf16(a, b, c.v);
    }
#pragma unroll
    for (int r = 0; r < 8; ++r) {
      float v = c.f[r];
      t2r[wave][(r + mhi) * 64 + n] = (bf16)(v > 0.f ? v : 0.f);
    }
  }
  // layer 3: 64 -> 1 dot + sigmoid, lanes 0..15 handle one row each
  if (lane < 16) {
    float acc = tb3[0];
#pragma unroll 8
    for (int k = 0; k < 64; ++k)
      acc += (float)t2r[wave][lane * 64 + k] * tw3[k];
    out[rowBase + lane] = 1.f / (1.f + __expf(-acc));
  }
}

extern "C" void kernel_launch(void* const* d_in, const int* in_sizes, int n_in,
                              void* d_out, int out_size, void* d_ws, size_t ws_size,
                              hipStream_t stream) {
  (void)in_sizes; (void)n_in; (void)out_size; (void)ws_size;
  const float* dense_x  = (const float*)d_in[0];
  const int*   sparse_x = (const int*)d_in[1];
  const float* emb      = (const float*)d_in[2];
  const float* bw1 = (const float*)d_in[3];
  const float* bb1 = (const float*)d_in[4];
  const float* bw2 = (const float*)d_in[5];
  const float* bb2 = (const float*)d_in[6];
  const float* tw1 = (const float*)d_in[7];
  const float* tb1 = (const float*)d_in[8];
  const float* tw2 = (const float*)d_in[9];
  const float* tb2 = (const float*)d_in[10];
  const float* tw3 = (const float*)d_in[11];
  const float* tb3 = (const float*)d_in[12];

  bf16* xfinal = (bf16*)d_ws;                       // B x 480 bf16 (~31.5 MB)
  bf16* W1f = xfinal + (size_t)B_TOTAL * KP;        // 16*512
  bf16* W2f = W1f + 16 * 512;                       // 64*512
  bf16* T1f = W2f + 64 * 512;                       // 120*512
  bf16* T2f = T1f + 120 * 512;                      // 16*512
  float* out = (float*)d_out;

  dlrm_pack_weights<<<64, 256, 0, stream>>>(bw1, bw2, tw1, tw2, W1f, W2f, T1f, T2f);
  dlrm_bottom_mlp<<<B_TOTAL / 64, 128, 0, stream>>>(dense_x, W1f, bb1, W2f, bb2, xfinal);
  dlrm_interact  <<<B_TOTAL / 2, 64, 0, stream>>>(sparse_x, emb, xfinal);
  dlrm_top_mlp   <<<B_TOTAL / 32, 64, 0, stream>>>(xfinal, T1f, tb1, T2f, tb2, tw3, tb3, out);
}
